// TBCNNEncoder_2439541424850
// MI455X (gfx1250) — matrix operand order repaired and checked by reading, not compiled, not run on hardware
//
#include <hip/hip_runtime.h>
#include <hip/hip_bf16.h>
#include <math.h>

typedef __attribute__((ext_vector_type(16))) _Float16 v16h;
typedef __attribute__((ext_vector_type(8)))  _Float16 v8h;
typedef __attribute__((ext_vector_type(8)))  float    v8f;

#define BB 16
#define NN 1024
#define CC 16
#define SS 8
#define BN (BB * NN)          // 16384 nodes
#define DIM 256               // D == O == 256
#define KCONV 768             // 3 * D contraction for conv layers

// ---------------------------------------------------------------------------
// Weight preparation: cast + transpose to f16 [O][K] so WMMA B-fragments are
// contiguous per lane (two b128 loads).
// ---------------------------------------------------------------------------
__global__ __launch_bounds__(256)
void prep_weights_kernel(const float* __restrict__ Wp,     // [256][256] (k,o)
                         const float* __restrict__ convW,  // [2][768][256] (l,kd,o)
                         _Float16* __restrict__ WpT,       // [256][256] (o,k)
                         _Float16* __restrict__ convWT)    // [2][256][768] (l,o,kd)
{
    int i = blockIdx.x * 256 + threadIdx.x;
    if (i < 256 * 256) {
        int o = i >> 8, k = i & 255;
        WpT[o * 256 + k] = (_Float16)Wp[k * 256 + o];
    } else {
        int j = i - 256 * 256;
        if (j < 2 * 256 * KCONV) {
            int l  = j / (256 * KCONV);
            int r  = j % (256 * KCONV);
            int o  = r / KCONV;
            int kd = r % KCONV;
            convWT[(size_t)l * 256 * KCONV + (size_t)o * KCONV + kd] =
                (_Float16)convW[(size_t)l * KCONV * 256 + (size_t)kd * 256 + o];
        }
    }
}

// ---------------------------------------------------------------------------
// Embedding: x[n] = concat(type_table[type[n]], sum_s token_table[tok[n][s]])
// one block (128 lanes over feature dim) per node -> coalesced gathers.
// ---------------------------------------------------------------------------
__global__ __launch_bounds__(128)
void embed_kernel(const int* __restrict__ node_type,
                  const int* __restrict__ node_tokens,
                  const float* __restrict__ type_table,
                  const float* __restrict__ token_table,
                  _Float16* __restrict__ xH)               // [BN][256]
{
    int n = blockIdx.x;
    int t = threadIdx.x;                                   // 0..127
    int ty = node_type[n];
    float te = type_table[(size_t)ty * 128 + t];
    float se = 0.f;
#pragma unroll
    for (int s = 0; s < SS; ++s) {
        int tok = node_tokens[n * SS + s];
        se += token_table[(size_t)tok * 128 + t];
    }
    xH[(size_t)n * 256 + t]       = (_Float16)te;
    xH[(size_t)n * 256 + 128 + t] = (_Float16)se;
}

// ---------------------------------------------------------------------------
// TBCNN positional coefficients for the 16 children (parent slot is trivial:
// eta_t = delta(parent), eta_r/eta_l parent entries are 0).
// ---------------------------------------------------------------------------
__global__ __launch_bounds__(256)
void coef_kernel(const int* __restrict__ children,         // [BN][16]
                 float* __restrict__ coefR,                // [BN][16]
                 float* __restrict__ coefL)                // [BN][16]
{
    int n = blockIdx.x * 256 + threadIdx.x;
    if (n >= BN) return;
    int cm[CC];
    int ns = 0;
#pragma unroll
    for (int c = 0; c < CC; ++c) {
        cm[c] = (children[n * CC + c] > 0) ? 1 : 0;
        ns += cm[c];
    }
    float denom = (float)((ns - 1) > 1 ? (ns - 1) : 1);
#pragma unroll
    for (int c = 0; c < CC; ++c) {
        float er;
        if (ns == 1) er = (c == 0) ? 0.5f : 0.0f;           // "singles" branch
        else         er = cm[c] ? ((float)c) / denom : 0.0f;
        float el = (1.0f - er) * (float)cm[c];
        coefR[n * CC + c] = er;
        coefL[n * CC + c] = el;
    }
}

// ---------------------------------------------------------------------------
// Mix: mixed[n] = [parent_vec, sum_c etaR*child_c, sum_c etaL*child_c]
// block = one node, 256 lanes over feature dim; child index uniform per block
// (scalar loads), vector gathers coalesced.
// ---------------------------------------------------------------------------
__global__ __launch_bounds__(256)
void mix_kernel(const _Float16* __restrict__ nodesH,       // [BN][256]
                const int* __restrict__ children,          // [BN][16]
                const float* __restrict__ coefR,
                const float* __restrict__ coefL,
                _Float16* __restrict__ mixed)              // [BN][768]
{
    int n = blockIdx.x;
    int d = threadIdx.x;
    int b = n / NN;
    float p  = (float)nodesH[(size_t)n * 256 + d];
    float aR = 0.f, aL = 0.f;
#pragma unroll 4
    for (int c = 0; c < CC; ++c) {
        int idx = children[n * CC + c];                    // uniform in block
        float v = 0.f;
        if (idx != 0)                                      // idx 0 => no child
            v = (float)nodesH[((size_t)(b * NN + idx)) * 256 + d];
        aR += coefR[n * CC + c] * v;
        aL += coefL[n * CC + c] * v;
    }
    mixed[(size_t)n * KCONV + d]        = (_Float16)p;
    mixed[(size_t)n * KCONV + 256 + d]  = (_Float16)aR;
    mixed[(size_t)n * KCONV + 512 + d]  = (_Float16)aL;
}

// ---------------------------------------------------------------------------
// WMMA GEMM: out[M x 256] = act(A[M x K] * Bt^T + bias)
//   A row-major [M][K] f16, Bt = weights pre-transposed [256][K] f16.
// One wave -> 16x64 output slab (4 f32 accumulators); 4 waves/block -> N=256.
// Software-pipelined: K-step i+1's A fragment and all four B fragments are
// loaded into fresh registers while the four WMMAs for K-step i execute, so
// the compiler can batch the b128 loads and avoid s_wait_loadcnt 0 before
// every WMMA.
// Fragment layouts used here: A(16x32 f16): lane<16 holds M=lane, K{0..15};
// lane>=16 holds K{16..31}. B(32x16 f16): lane holds col (lane&15),
// 16 contiguous K starting at (lane>=16 ? 16 : 0) from the transposed row.
// ---------------------------------------------------------------------------
typedef union { v16h v; v8h h[2]; } Frag;

#define WMMA_F16(a, b, c) __builtin_amdgcn_wmma_f32_16x16x32_f16( \
        false, (a), false, (b), (short)0, (c), false, false)

__global__ __launch_bounds__(128)
void gemm_wmma_kernel(const _Float16* __restrict__ A,
                      const _Float16* __restrict__ Bt,
                      const float* __restrict__ bias,
                      _Float16* __restrict__ outH,         // nullable
                      float* __restrict__ outF,            // nullable
                      int K, int doTanh)
{
    int tid  = threadIdx.x;
    int wave = tid >> 5;
    int lane = tid & 31;
    int rowBase = blockIdx.x * 16;
    int colBase = wave * 64;
    int laneLo  = lane & 15;
    int hiHalf  = lane >> 4;
    int aKoff   = hiHalf << 4;    // 0 or 16 halves
    int bKoff   = hiHalf << 4;    // 0 or 16 halves

    const _Float16* arow  = A  + (size_t)(rowBase + laneLo) * K + aKoff;
    const _Float16* brow0 = Bt + (size_t)(colBase +  0 + laneLo) * K + bKoff;
    const _Float16* brow1 = Bt + (size_t)(colBase + 16 + laneLo) * K + bKoff;
    const _Float16* brow2 = Bt + (size_t)(colBase + 32 + laneLo) * K + bKoff;
    const _Float16* brow3 = Bt + (size_t)(colBase + 48 + laneLo) * K + bKoff;

    v8f acc0 = {}, acc1 = {}, acc2 = {}, acc3 = {};

    // Prologue: K-step 0 fragments.
    Frag a, b0, b1, b2, b3;
    a.h[0]  = *(const v8h*)(arow);       a.h[1]  = *(const v8h*)(arow + 8);
    b0.h[0] = *(const v8h*)(brow0);      b0.h[1] = *(const v8h*)(brow0 + 8);
    b1.h[0] = *(const v8h*)(brow1);      b1.h[1] = *(const v8h*)(brow1 + 8);
    b2.h[0] = *(const v8h*)(brow2);      b2.h[1] = *(const v8h*)(brow2 + 8);
    b3.h[0] = *(const v8h*)(brow3);      b3.h[1] = *(const v8h*)(brow3 + 8);

#pragma unroll 2
    for (int kb = 32; kb < K; kb += 32) {
        Frag an, bn0, bn1, bn2, bn3;     // prefetch next K-step
        an.h[0]  = *(const v8h*)(arow + kb);       an.h[1]  = *(const v8h*)(arow + kb + 8);
        bn0.h[0] = *(const v8h*)(brow0 + kb);      bn0.h[1] = *(const v8h*)(brow0 + kb + 8);
        bn1.h[0] = *(const v8h*)(brow1 + kb);      bn1.h[1] = *(const v8h*)(brow1 + kb + 8);
        bn2.h[0] = *(const v8h*)(brow2 + kb);      bn2.h[1] = *(const v8h*)(brow2 + kb + 8);
        bn3.h[0] = *(const v8h*)(brow3 + kb);      bn3.h[1] = *(const v8h*)(brow3 + kb + 8);

        acc0 = WMMA_F16(a.v, b0.v, acc0);          // compute on current step
        acc1 = WMMA_F16(a.v, b1.v, acc1);
        acc2 = WMMA_F16(a.v, b2.v, acc2);
        acc3 = WMMA_F16(a.v, b3.v, acc3);

        a = an; b0 = bn0; b1 = bn1; b2 = bn2; b3 = bn3;  // rotate (unroll-2 -> renames)
    }
    // Epilogue: last K-step.
    acc0 = WMMA_F16(a.v, b0.v, acc0);
    acc1 = WMMA_F16(a.v, b1.v, acc1);
    acc2 = WMMA_F16(a.v, b2.v, acc2);
    acc3 = WMMA_F16(a.v, b3.v, acc3);

    // C/D layout: VGPR j holds row rowBase + j (+8 for lanes>=16), col lane&15.
    int rOff = hiHalf << 3;
    v8f accs[4] = { acc0, acc1, acc2, acc3 };
#pragma unroll
    for (int t = 0; t < 4; ++t) {
        int col = colBase + t * 16 + laneLo;
        float bv = bias[col];
#pragma unroll
        for (int j = 0; j < 8; ++j) {
            int row = rowBase + rOff + j;
            float v = accs[t][j] + bv;
            if (doTanh) v = tanhf(v);
            size_t o = (size_t)row * 256 + col;
            if (outH) outH[o] = (_Float16)v;
            if (outF) outF[o] = v;
        }
    }
}

// ---------------------------------------------------------------------------
// Attention score per node: wt[n] = mask ? sigmoid(nodes[n].attn_a) : 0
// one wave per node, shuffle reduction.
// ---------------------------------------------------------------------------
__global__ __launch_bounds__(256)
void score_kernel(const float* __restrict__ nodesF,        // [BN][256]
                  const int* __restrict__ node_type,
                  const float* __restrict__ attn_a,
                  float* __restrict__ wt)                  // [BN]
{
    int gtid = blockIdx.x * 256 + threadIdx.x;
    int n    = gtid >> 5;
    int lane = gtid & 31;
    if (n >= BN) return;
    float s = 0.f;
#pragma unroll
    for (int o = lane; o < 256; o += 32)
        s += nodesF[(size_t)n * 256 + o] * attn_a[o];
#pragma unroll
    for (int off = 16; off > 0; off >>= 1)
        s += __shfl_xor(s, off, 32);
    if (lane == 0)
        wt[n] = (node_type[n] != 0) ? (1.0f / (1.0f + expf(-s))) : 0.0f;
}

// ---------------------------------------------------------------------------
// code_vector[b] = mean_n nodes[b,n,:] * wt[b,n]   (mask folded into wt)
// ---------------------------------------------------------------------------
__global__ __launch_bounds__(256)
void agg_kernel(const float* __restrict__ nodesF,
                const float* __restrict__ wt,
                float* __restrict__ code)                  // [B][256]
{
    int b = blockIdx.x;
    int o = threadIdx.x;
    float acc = 0.f;
    for (int n = 0; n < NN; ++n)
        acc += nodesF[((size_t)b * NN + n) * 256 + o] * wt[b * NN + n];
    code[b * 256 + o] = acc * (1.0f / (float)NN);
}

// ---------------------------------------------------------------------------
extern "C" void kernel_launch(void* const* d_in, const int* in_sizes, int n_in,
                              void* d_out, int out_size, void* d_ws, size_t ws_size,
                              hipStream_t stream) {
    const int*   node_type   = (const int*)d_in[0];        // [B,N]
    const int*   node_tokens = (const int*)d_in[1];        // [B,N,S]
    const int*   children    = (const int*)d_in[2];        // [B,N,C]
    const float* type_table  = (const float*)d_in[3];      // [200,128]
    const float* token_table = (const float*)d_in[4];      // [50000,128]
    const float* W_parent    = (const float*)d_in[5];      // [256,256]
    const float* b_parent    = (const float*)d_in[6];      // [256]
    const float* conv_w      = (const float*)d_in[7];      // [2,3,256,256]
    const float* conv_b      = (const float*)d_in[8];      // [2,256]
    const float* attn_a      = (const float*)d_in[9];      // [256]

    char* ws = (char*)d_ws;
    size_t off = 0;
    _Float16* WpT     = (_Float16*)(ws + off); off += (size_t)256 * 256 * 2;          // 128 KB
    _Float16* convWT  = (_Float16*)(ws + off); off += (size_t)2 * 256 * KCONV * 2;    // 768 KB
    _Float16* xH      = (_Float16*)(ws + off); off += (size_t)BN * 256 * 2;           // 8 MB
    _Float16* nodesH0 = (_Float16*)(ws + off); off += (size_t)BN * 256 * 2;           // 8 MB
    _Float16* nodesH1 = (_Float16*)(ws + off); off += (size_t)BN * 256 * 2;           // 8 MB
    _Float16* mixedH  = (_Float16*)(ws + off); off += (size_t)BN * KCONV * 2;         // 24 MB
    float*    coefR   = (float*)(ws + off);    off += (size_t)BN * CC * 4;            // 1 MB
    float*    coefL   = (float*)(ws + off);    off += (size_t)BN * CC * 4;            // 1 MB
    float*    wt      = (float*)(ws + off);    off += (size_t)BN * 4;                 // 64 KB

    float* code   = (float*)d_out;                         // [B,256]
    float* nodesF = (float*)d_out + (size_t)BB * 256;      // [B,N,256]

    // 1) weight cast+transpose
    {
        int total = 256 * 256 + 2 * 256 * KCONV;
        prep_weights_kernel<<<(total + 255) / 256, 256, 0, stream>>>(
            W_parent, conv_w, WpT, convWT);
    }
    // 2) embeddings
    embed_kernel<<<BN, 128, 0, stream>>>(node_type, node_tokens,
                                         type_table, token_table, xH);
    // 3) positional coefficients
    coef_kernel<<<BN / 256, 256, 0, stream>>>(children, coefR, coefL);
    // 4) parent projection (WMMA, K=256)
    gemm_wmma_kernel<<<BN / 16, 128, 0, stream>>>(
        xH, WpT, b_parent, nodesH0, (float*)nullptr, 256, 0);
    // 5) layer 0: mix + WMMA GEMM + tanh (K=768)
    mix_kernel<<<BN, 256, 0, stream>>>(nodesH0, children, coefR, coefL, mixedH);
    gemm_wmma_kernel<<<BN / 16, 128, 0, stream>>>(
        mixedH, convWT, conv_b, nodesH1, (float*)nullptr, KCONV, 1);
    // 6) layer 1: mix + WMMA GEMM + tanh, f32 nodes straight into d_out
    mix_kernel<<<BN, 256, 0, stream>>>(nodesH1, children, coefR, coefL, mixedH);
    gemm_wmma_kernel<<<BN / 16, 128, 0, stream>>>(
        mixedH, convWT + (size_t)256 * KCONV, conv_b + 256,
        (_Float16*)nullptr, nodesF, KCONV, 1);
    // 7) attention weights + weighted mean
    score_kernel<<<(BN * 32) / 256, 256, 0, stream>>>(nodesF, node_type, attn_a, wt);
    agg_kernel<<<BB, 256, 0, stream>>>(nodesF, wt, code);
}
